// Model_52776558133451
// MI455X (gfx1250) — matrix operand order repaired and checked by reading, not compiled, not run on hardware
//
#include <hip/hip_runtime.h>
#include <math.h>

// ---------------- model constants ----------------
#define TSN   600
#define MBN   50
#define CTSN  80
#define NHID  400
#define NATT  10
#define NDEN  121
#define ROWS  (TSN*MBN)      // 30000
#define ROWS4 30016          // rows padded to multiple of 64 (4 M-tiles)
#define KP    416            // K padded to 13*32
#define KT13  13
#define MBP   64             // batch rows padded to 4 M-tiles
#define NWG   32             // persistent workgroups in seq kernel
#define TPB   256

typedef unsigned short u16;
typedef __bf16 bf16_t;
typedef bf16_t v16bf __attribute__((ext_vector_type(16)));
typedef float  v8f   __attribute__((ext_vector_type(8)));

struct alignas(16) B128 { unsigned u[4]; };
union AF { B128 h[2]; v16bf v; };

__device__ inline float bf2f(u16 h){
  union { unsigned u; float f; } c; c.u = ((unsigned)h) << 16; return c.f;
}
__device__ inline u16 f2bf(float f){
  union { float f; unsigned u; } c; c.f = f;
  unsigned r = ((c.u >> 16) & 1u) + 0x7fffu;
  return (u16)((c.u + r) >> 16);
}
__device__ inline float sigmoidf_(float x){ return 1.f / (1.f + __expf(-x)); }

// ---------------- WMMA fragment helpers (wave32 layouts, ISA 7.12.2) ----------------
// A (16x32 bf16): lane L holds row m0+(L&15); elems 0..7 = K[half*8..+7], 8..15 = K[16+half*8..+7]
__device__ inline v16bf load_afrag(const u16* A, int lda, int m0, int kt, int lane){
  int m = m0 + (lane & 15);
  int half = lane >> 4;
  const u16* p = A + (size_t)m * lda + kt * 32 + half * 8;
  AF f;
  f.h[0] = *(const B128*)p;
  f.h[1] = *(const B128*)(p + 16);
  return f.v;
}
// B packed layout: [nt][kt][lane][16 bf16] ; lane holds N=lane&15, K=(i>>3)*16 + (lane>>4)*8 + (i&7)
__device__ inline v16bf load_bfrag(const u16* Bpk, int nt, int kt, int lane){
  const u16* p = Bpk + (((size_t)nt * KT13 + kt) * 32 + lane) * 16;
  AF f;
  f.h[0] = *(const B128*)p;
  f.h[1] = *(const B128*)(p + 8);
  return f.v;
}
__device__ inline v8f wmma_bf(v16bf a, v16bf b, v8f c){
  return __builtin_amdgcn_wmma_f32_16x16x32_bf16(false, a, false, b, (short)0, c, false, false);
}

// ---------------- weight packing: f32 (K,N) row-major -> bf16 B-fragment layout ----------------
__global__ __launch_bounds__(256) void pack_kernel(const float* W, u16* dst, int K, int N, int total){
  int id = blockIdx.x * 256 + threadIdx.x;
  if (id >= total) return;
  int i    = id & 15;
  int lane = (id >> 4) & 31;
  int tk   = id >> 9;          // nt*KT13 + kt
  int kt   = tk % KT13;
  int nt   = tk / KT13;
  int n = nt * 16 + (lane & 15);
  int k = kt * 32 + (i >> 3) * 16 + (lane >> 4) * 8 + (i & 7);
  float v = (k < K && n < N) ? W[(size_t)k * N + n] : 0.f;
  dst[id] = f2bf(v);
}

__global__ void obias_kernel(const float* b1, const float* b2, const float* b3, float* ob){
  int j = threadIdx.x;
  if (j < 128) ob[j] = (j < NDEN) ? (b1[j] + b2[j] + b3[j]) : 0.f;
}

// x = inp @ lproj_W + lproj_b  (K=3), stored bf16 in A-row layout (pad K to 416)
__global__ __launch_bounds__(256) void xproj_kernel(const float* inp, const float* lW, const float* lb,
                                                    u16* xbf){
  size_t id = (size_t)blockIdx.x * 256 + threadIdx.x;
  if (id >= (size_t)ROWS * KP) return;
  int row = (int)(id / KP), h = (int)(id - (size_t)row * KP);
  float v = 0.f;
  if (h < NHID){
    v = lb[h] + inp[row*3+0]*lW[h] + inp[row*3+1]*lW[NHID+h] + inp[row*3+2]*lW[2*NHID+h];
  }
  xbf[id] = f2bf(v);
}

// attention heads: exp(x @ W + b) for a, b, k
__global__ __launch_bounds__(256) void heads_kernel(const u16* xbf,
                                                    const float* Wa, const float* ba,
                                                    const float* Wb, const float* bb,
                                                    const float* Wk, const float* bk,
                                                    float* aexp, float* bexp, float* kex){
  int id = blockIdx.x * 256 + threadIdx.x;
  if (id >= ROWS * NATT) return;
  int row = id / NATT, j = id - row * NATT;
  float a = ba[j], b = bb[j], k = bk[j];
  const u16* xr = xbf + (size_t)row * KP;
  for (int h = 0; h < NHID; ++h){
    float xv = bf2f(xr[h]);
    a = fmaf(xv, Wa[h*NATT+j], a);
    b = fmaf(xv, Wb[h*NATT+j], b);
    k = fmaf(xv, Wk[h*NATT+j], k);
  }
  aexp[id] = __expf(a);
  bexp[id] = __expf(b);
  kex[id]  = __expf(k);
}

// cumulative sum of kex over t (500 independent chains)
__global__ void kcum_kernel(const float* kinit, float* kex){
  int id = threadIdx.x;
  if (id >= MBN * NATT) return;
  int b = id / NATT, j = id - b * NATT;
  float k = kinit[id];
  for (int t = 0; t < TSN; ++t){
    size_t idx = ((size_t)t * MBN + b) * NATT + j;
    k += kex[idx];
    kex[idx] = k;
  }
}

// generic precompute GEMM, 4-Mtile register blocking:
// out = A1@B1 (+ A2@B2) + bias1 (+ bias2), bf16 or f32 store. rows4 % 64 == 0.
__global__ __launch_bounds__(256) void gemm_pre(const u16* A1, const u16* B1,
                                                const u16* A2, const u16* B2,
                                                const float* bias1, const float* bias2,
                                                u16* obf, float* of32,
                                                int rows4, int NT, int ldc){
  int lane = threadIdx.x & 31;
  int wid  = (blockIdx.x * blockDim.x + threadIdx.x) >> 5;
  int nw   = (gridDim.x * blockDim.x) >> 5;
  int MT4 = rows4 >> 6;
  long total = (long)MT4 * NT;
  for (long u = wid; u < total; u += nw){
    int nt = (int)(u % NT);
    int m0 = (int)(u / NT) << 6;
    __builtin_prefetch(B1 + (size_t)nt * KT13 * 512);
    v8f c[4];
    #pragma unroll
    for (int mt = 0; mt < 4; ++mt)
      for (int i = 0; i < 8; ++i) c[mt][i] = 0.f;
    for (int kt = 0; kt < KT13; ++kt){
      v16bf b = load_bfrag(B1, nt, kt, lane);
      #pragma unroll
      for (int mt = 0; mt < 4; ++mt){
        v16bf a = load_afrag(A1, KP, m0 + mt*16, kt, lane);
        c[mt] = wmma_bf(a, b, c[mt]);
      }
    }
    if (A2){
      for (int kt = 0; kt < KT13; ++kt){
        v16bf b = load_bfrag(B2, nt, kt, lane);
        #pragma unroll
        for (int mt = 0; mt < 4; ++mt){
          v16bf a = load_afrag(A2, KP, m0 + mt*16, kt, lane);
          c[mt] = wmma_bf(a, b, c[mt]);
        }
      }
    }
    int n = nt * 16 + (lane & 15);
    float badd = (bias1 ? bias1[n] : 0.f) + (bias2 ? bias2[n] : 0.f);
    #pragma unroll
    for (int mt = 0; mt < 4; ++mt){
      int mb = m0 + mt*16 + (lane >> 4) * 8;
      for (int i = 0; i < 8; ++i){
        size_t idx = (size_t)(mb + i) * ldc + n;
        float v = c[mt][i] + badd;
        if (obf) obf[idx] = f2bf(v);
        else     of32[idx] = v;
      }
    }
  }
}

// attention window: phi + w = phi@ctx ; writes w (A-row bf16) and w+finp (A-row bf16)
__global__ __launch_bounds__(128) void attn_kernel(const int* c_inp, const float* emb,
                                                   const float* aexp, const float* bexp, const float* kcum,
                                                   const u16* finp, u16* wbf, u16* wfbf){
  int row = blockIdx.x;             // t*MB + b
  int b   = row % MBN;
  __shared__ float phi[CTSN];
  __shared__ float sa[NATT], sb[NATT], sk[NATT];
  int tid = threadIdx.x;
  if (tid < NATT){
    sa[tid] = aexp[(size_t)row*NATT + tid];
    sb[tid] = bexp[(size_t)row*NATT + tid];
    sk[tid] = kcum[(size_t)row*NATT + tid];
  }
  __syncthreads();
  if (tid < CTSN){
    float u = (float)tid, s = 0.f;
    for (int j = 0; j < NATT; ++j){
      float d = sk[j] - u;
      s = fmaf(sa[j], __expf(-sb[j] * d * d), s);
    }
    phi[tid] = s;
  }
  __syncthreads();
  for (int h = tid; h < KP; h += 128){
    if (h < NHID){
      float acc = 0.f;
      for (int c = 0; c < CTSN; ++c)
        acc = fmaf(phi[c], emb[(size_t)c_inp[c*MBN + b] * NHID + h], acc);
      wbf [(size_t)row*KP + h] = f2bf(acc);
      wfbf[(size_t)row*KP + h] = f2bf(acc + bf2f(finp[(size_t)row*NHID + h]));
    } else {
      wbf [(size_t)row*KP + h] = 0;
      wfbf[(size_t)row*KP + h] = 0;
    }
  }
}

// ---------------- persistent sequence kernel ----------------
struct SeqArgs {
  const u16 *pk_u1, *pk_u2, *pk_u3, *pk_wur1, *pk_wur2, *pk_wur3;
  const u16 *pk_l12, *pk_l13, *pk_l23, *pk_lo1, *pk_lo2, *pk_lo3;
  const u16 *F1, *f2p, *f3p;
  const float *l12b, *l13b, *l23b;
  const float *poutW, *poutB, *obias, *mask;
  const float *h1i, *h2i, *h3i;
  u16 *h1, *h2, *h3, *A1, *A2, *A3;
  float *gacc1, *gacc2, *gacc3, *upd1, *upd2, *upd3, *f2buf, *f3buf, *outacc;
  float *dout;
  unsigned *bar;
};

struct Job {
  const u16* A; const u16* B; int NT; int mode;  // 0=store, 1=accum, 2=gru
  float* C; int ldc;
  const u16* cb;     // bf16 C-init (ld=ldc) / GRU1 state base (ld=1200)
  const float* sf;   // f32 state base (ld=1200) for GRU2/3
  const float* upd;  // update gate (ld=NHID)
  u16* hbuf;         // h state A-row buffer (ld=KP)
  const float* ob;   // per-column bias
  const float* mrow; // mask row
};

__device__ inline Job jget(const u16* A, const u16* B, int NT, int mode, float* C, int ldc,
                           const u16* cb, const float* sf, const float* upd, u16* hbuf,
                           const float* ob, const float* mrow){
  Job j; j.A=A; j.B=B; j.NT=NT; j.mode=mode; j.C=C; j.ldc=ldc;
  j.cb=cb; j.sf=sf; j.upd=upd; j.hbuf=hbuf; j.ob=ob; j.mrow=mrow; return j;
}

__device__ inline void gbar(unsigned* bar){
  __threadfence();
  __syncthreads();
  if (threadIdx.x == 0){
    unsigned gen = __hip_atomic_load(bar + 1, __ATOMIC_RELAXED, __HIP_MEMORY_SCOPE_AGENT);
    unsigned arrived = __hip_atomic_fetch_add(bar, 1u, __ATOMIC_ACQ_REL, __HIP_MEMORY_SCOPE_AGENT);
    if (arrived == (unsigned)(gridDim.x - 1)){
      __hip_atomic_store(bar, 0u, __ATOMIC_RELAXED, __HIP_MEMORY_SCOPE_AGENT);
      __hip_atomic_fetch_add(bar + 1, 1u, __ATOMIC_RELEASE, __HIP_MEMORY_SCOPE_AGENT);
    } else {
      while (__hip_atomic_load(bar + 1, __ATOMIC_ACQUIRE, __HIP_MEMORY_SCOPE_AGENT) == gen)
        __builtin_amdgcn_s_sleep(2);
    }
  }
  __syncthreads();
  __threadfence();
}

// one work unit = one 16-wide N tile across all 4 M tiles (B fragment reused 4x)
__device__ void run_jobs(const Job* js, int nj, int wid, int nw, int lane){
  int start[4]; int total = 0;
  for (int q = 0; q < nj; ++q){ start[q] = total; total += js[q].NT; }
  for (int unit = wid; unit < total; unit += nw){
    int q = nj - 1;
    while (q > 0 && unit < start[q]) --q;
    const Job J = js[q];
    int nt = unit - start[q];
    __builtin_prefetch(J.B + (size_t)nt * KT13 * 512);
    v8f c[4];
    #pragma unroll
    for (int mt = 0; mt < 4; ++mt)
      for (int i = 0; i < 8; ++i) c[mt][i] = 0.f;
    for (int kt = 0; kt < KT13; ++kt){
      v16bf b = load_bfrag(J.B, nt, kt, lane);
      #pragma unroll
      for (int mt = 0; mt < 4; ++mt){
        v16bf a = load_afrag(J.A, KP, mt*16, kt, lane);
        c[mt] = wmma_bf(a, b, c[mt]);
      }
    }
    int n = nt * 16 + (lane & 15);
    #pragma unroll
    for (int mt = 0; mt < 4; ++mt){
      int mb = mt*16 + (lane >> 4) * 8;
      if (J.mode == 2){
        for (int i = 0; i < 8; ++i){
          int m = mb + i; if (m >= MBN) continue;
          float st = J.sf ? J.sf[(size_t)m*1200 + n] : bf2f(J.cb[(size_t)m*1200 + n]);
          float nx = tanhf(c[mt][i] + st);
          float u  = J.upd[m*NHID + n];
          float p  = bf2f(J.hbuf[m*KP + n]);
          float nf = nx + u + (p + (1.f - u));      // as written in reference _ggru
          float mk = J.mrow[m];
          J.hbuf[m*KP + n] = f2bf(mk * nf + (1.f - mk) * p);
        }
      } else {
        for (int i = 0; i < 8; ++i){
          int m = mb + i; if (m >= MBN) continue;
          float v = c[mt][i];
          if (J.ob) v += J.ob[n];
          if (J.cb) v += bf2f(J.cb[(size_t)m*J.ldc + n]);
          if (J.mode == 0) J.C[(size_t)m*J.ldc + n] = v;
          else             J.C[(size_t)m*J.ldc + n] += v;
        }
      }
    }
  }
}

// gate elementwise: upd/reset from gacc + gate_inp; A = state * reset (bf16 A-row)
__device__ inline void ephase(const float* gacc, const u16* fb, const float* ff,
                              float* upd, u16* Ab, int tid, int nthr){
  for (int id = tid; id < MBN * NHID; id += nthr){
    int m = id / NHID, k = id - m * NHID;
    float st, gu, gr;
    if (fb){
      const u16* r = fb + (size_t)m * 1200;
      st = bf2f(r[k]); gu = bf2f(r[400 + k]); gr = bf2f(r[800 + k]);
    } else {
      const float* r = ff + (size_t)m * 1200;
      st = r[k]; gu = r[400 + k]; gr = r[800 + k];
    }
    float u  = sigmoidf_(gacc[m*800 + k] + gu);
    float rr = sigmoidf_(gacc[m*800 + 400 + k] + gr);
    upd[id] = u;
    Ab[m*KP + k] = f2bf(st * rr);
  }
}

__device__ void finalphase(const SeqArgs& S, int t, int tid, int nthr){
  for (int id = tid; id < MBN * NDEN; id += nthr){
    int m = id / NDEN, j = id - m * NDEN;
    float acc = S.poutB[j];
    const float* oa = S.outacc + (size_t)m * 128;
    for (int i = 0; i < NDEN; ++i) acc = fmaf(oa[i], S.poutW[i*NDEN + j], acc);
    size_t r = (size_t)t * MBN + m;
    if      (j < 40)  S.dout[r*40 + j] = acc;                                        // mu
    else if (j < 80)  S.dout[(size_t)1200000 + r*40 + (j-40)] = __expf(acc) + 1e-4f; // sigma
    else if (j < 100) S.dout[(size_t)2400000 + r*20 + (j-80)] = tanhf(acc);          // corr
    else if (j < 120) S.dout[(size_t)3000000 + r*20 + (j-100)] = acc;                // log_coeff
    else              S.dout[(size_t)3600000 + r] = acc;                             // log_binary
  }
  if (t + 1 < TSN)
    ephase(S.gacc1, S.F1 + (size_t)(t+1) * MBN * 1200, nullptr, S.upd1, S.A1, tid, nthr);
}

__global__ __launch_bounds__(TPB, 1) void seq_kernel(SeqArgs S){
  const int tid  = blockIdx.x * TPB + threadIdx.x;
  const int nthr = gridDim.x * TPB;
  const int lane = threadIdx.x & 31;
  const int wid  = tid >> 5;
  const int nw   = nthr >> 5;

  // init state buffers (pads -> 0)
  for (int id = tid; id < MBP * KP; id += nthr){
    int m = id / KP, k = id - m * KP;
    u16 a = 0, b = 0, c = 0;
    if (m < MBN && k < NHID){
      a = f2bf(S.h1i[m*NHID + k]);
      b = f2bf(S.h2i[m*NHID + k]);
      c = f2bf(S.h3i[m*NHID + k]);
    }
    S.h1[id] = a; S.h2[id] = b; S.h3[id] = c;
    S.A1[id] = 0; S.A2[id] = 0; S.A3[id] = 0;
  }
  gbar(S.bar);

  { // gate accumulators for t = 0
    Job j[3];
    j[0] = jget(S.h1, S.pk_wur1, 50, 0, S.gacc1, 800, 0, 0, 0, 0, 0, 0);
    j[1] = jget(S.h2, S.pk_wur2, 50, 0, S.gacc2, 800, 0, 0, 0, 0, 0, 0);
    j[2] = jget(S.h3, S.pk_wur3, 50, 0, S.gacc3, 800, 0, 0, 0, 0, 0, 0);
    run_jobs(j, 3, wid, nw, lane);
  }
  gbar(S.bar);
  ephase(S.gacc1, S.F1, nullptr, S.upd1, S.A1, tid, nthr);
  gbar(S.bar);

  for (int t = 0; t < TSN; ++t){
    const float* mrow = S.mask + (size_t)t * MBN;
    { // GRU1: nxt = tanh(A1 @ att_U + state), h1 update in epilogue
      Job j = jget(S.A1, S.pk_u1, 25, 2, 0, 0, S.F1 + (size_t)t*MBN*1200, 0, S.upd1, S.h1, 0, mrow);
      run_jobs(&j, 1, wid, nw, lane);
    }
    gbar(S.bar);
    { // f2 = f2p + h1@l1_to_l2 + l12b ; f3 = f3p + h1@l1_to_l3 + l13b ; out = h1@loutp1 + obias
      Job j[3];
      j[0] = jget(S.h1, S.pk_l12, 75, 0, S.f2buf, 1200, S.f2p + (size_t)t*MBN*1200, 0, 0, 0, S.l12b, 0);
      j[1] = jget(S.h1, S.pk_l13, 75, 0, S.f3buf, 1200, S.f3p + (size_t)t*MBN*1200, 0, 0, 0, S.l13b, 0);
      j[2] = jget(S.h1, S.pk_lo1, 8, 0, S.outacc, 128, 0, 0, 0, 0, S.obias, 0);
      run_jobs(j, 3, wid, nw, lane);
    }
    gbar(S.bar);
    ephase(S.gacc2, nullptr, S.f2buf, S.upd2, S.A2, tid, nthr);
    gbar(S.bar);
    { Job j = jget(S.A2, S.pk_u2, 25, 2, 0, 0, 0, S.f2buf, S.upd2, S.h2, 0, mrow);
      run_jobs(&j, 1, wid, nw, lane); }
    gbar(S.bar);
    { // f3 += h2@l2_to_l3 + l23b ; out += h2@loutp2
      Job j[2];
      j[0] = jget(S.h2, S.pk_l23, 75, 1, S.f3buf, 1200, 0, 0, 0, 0, S.l23b, 0);
      j[1] = jget(S.h2, S.pk_lo2, 8, 1, S.outacc, 128, 0, 0, 0, 0, 0, 0);
      run_jobs(j, 2, wid, nw, lane);
    }
    gbar(S.bar);
    ephase(S.gacc3, nullptr, S.f3buf, S.upd3, S.A3, tid, nthr);
    gbar(S.bar);
    { Job j = jget(S.A3, S.pk_u3, 25, 2, 0, 0, 0, S.f3buf, S.upd3, S.h3, 0, mrow);
      run_jobs(&j, 1, wid, nw, lane); }
    gbar(S.bar);
    { // out += h3@loutp3 ; gate accumulators for t+1
      Job j[4];
      j[0] = jget(S.h3, S.pk_lo3, 8, 1, S.outacc, 128, 0, 0, 0, 0, 0, 0);
      j[1] = jget(S.h1, S.pk_wur1, 50, 0, S.gacc1, 800, 0, 0, 0, 0, 0, 0);
      j[2] = jget(S.h2, S.pk_wur2, 50, 0, S.gacc2, 800, 0, 0, 0, 0, 0, 0);
      j[3] = jget(S.h3, S.pk_wur3, 50, 0, S.gacc3, 800, 0, 0, 0, 0, 0, 0);
      run_jobs(j, 4, wid, nw, lane);
    }
    gbar(S.bar);
    finalphase(S, t, tid, nthr);   // poutp + nonlinearity + E1(t+1)
    gbar(S.bar);
  }
}

// ---------------- host launcher ----------------
extern "C" void kernel_launch(void* const* d_in, const int* in_sizes, int n_in,
                              void* d_out, int out_size, void* d_ws, size_t ws_size,
                              hipStream_t stream){
  (void)in_sizes; (void)n_in; (void)out_size; (void)ws_size;
  const int*   c_inp = (const int*)  d_in[0];
  const float* inp   = (const float*)d_in[1];
  const float* mask  = (const float*)d_in[2];
  const float* h1i   = (const float*)d_in[3];
  const float* kinit = (const float*)d_in[4];
  const float* h2i   = (const float*)d_in[5];
  const float* h3i   = (const float*)d_in[6];
  const float* emb   = (const float*)d_in[7];
  const float* lW    = (const float*)d_in[8];
  const float* lb    = (const float*)d_in[9];
  const float* aW  = (const float*)d_in[10]; const float* ab_ = (const float*)d_in[11];
  const float* bW  = (const float*)d_in[12]; const float* bb_ = (const float*)d_in[13];
  const float* kW  = (const float*)d_in[14]; const float* kb_ = (const float*)d_in[15];
  const float* ifW = (const float*)d_in[16]; const float* ifb = (const float*)d_in[17];
  const float* afW = (const float*)d_in[18]; const float* afb = (const float*)d_in[19];
  const float* p2W = (const float*)d_in[20]; const float* p2b = (const float*)d_in[21];
  const float* p3W = (const float*)d_in[22]; const float* p3b = (const float*)d_in[23];
  const float* a2W = (const float*)d_in[24]; const float* a2b = (const float*)d_in[25];
  const float* a3W = (const float*)d_in[26]; const float* a3b = (const float*)d_in[27];
  const float* l12W = (const float*)d_in[28]; const float* l12b = (const float*)d_in[29];
  const float* l13W = (const float*)d_in[30]; const float* l13b = (const float*)d_in[31];
  const float* l23W = (const float*)d_in[32]; const float* l23b = (const float*)d_in[33];
  const float* wur1W = (const float*)d_in[34]; const float* u1W = (const float*)d_in[35];
  const float* wur2W = (const float*)d_in[36]; const float* u2W = (const float*)d_in[37];
  const float* wur3W = (const float*)d_in[38]; const float* u3W = (const float*)d_in[39];
  const float* lo1W = (const float*)d_in[40]; const float* lo1b = (const float*)d_in[41];
  const float* lo2W = (const float*)d_in[42]; const float* lo2b = (const float*)d_in[43];
  const float* lo3W = (const float*)d_in[44]; const float* lo3b = (const float*)d_in[45];
  const float* poW  = (const float*)d_in[46]; const float* pob  = (const float*)d_in[47];

  char* w = (char*)d_ws;
  size_t off = 0;
  auto alloc = [&](size_t bytes)->char*{
    char* p = w + off; off += (bytes + 255) & ~(size_t)255; return p;
  };
  auto pkb = [](int NT)->size_t{ return (size_t)NT * KT13 * 512 * 2; };

  u16* pk_if  = (u16*)alloc(pkb(25));
  u16* pk_af  = (u16*)alloc(pkb(75));
  u16* pk_p2  = (u16*)alloc(pkb(75));
  u16* pk_a2  = (u16*)alloc(pkb(75));
  u16* pk_p3  = (u16*)alloc(pkb(75));
  u16* pk_a3  = (u16*)alloc(pkb(75));
  u16* pk_l12 = (u16*)alloc(pkb(75));
  u16* pk_l13 = (u16*)alloc(pkb(75));
  u16* pk_l23 = (u16*)alloc(pkb(75));
  u16* pk_wur1 = (u16*)alloc(pkb(50));
  u16* pk_wur2 = (u16*)alloc(pkb(50));
  u16* pk_wur3 = (u16*)alloc(pkb(50));
  u16* pk_u1 = (u16*)alloc(pkb(25));
  u16* pk_u2 = (u16*)alloc(pkb(25));
  u16* pk_u3 = (u16*)alloc(pkb(25));
  u16* pk_lo1 = (u16*)alloc(pkb(8));
  u16* pk_lo2 = (u16*)alloc(pkb(8));
  u16* pk_lo3 = (u16*)alloc(pkb(8));
  float* obias = (float*)alloc(128 * 4);
  u16* xbf  = (u16*)alloc((size_t)ROWS4 * KP * 2);
  u16* finp = (u16*)alloc((size_t)ROWS4 * NHID * 2);
  u16* wbf  = (u16*)alloc((size_t)ROWS4 * KP * 2);
  u16* wfbf = (u16*)alloc((size_t)ROWS4 * KP * 2);
  float* aexp = (float*)alloc((size_t)ROWS * NATT * 4);
  float* bexp = (float*)alloc((size_t)ROWS * NATT * 4);
  float* kex  = (float*)alloc((size_t)ROWS * NATT * 4);
  u16* F1  = (u16*)alloc((size_t)ROWS4 * 1200 * 2);
  u16* f2p = (u16*)alloc((size_t)ROWS4 * 1200 * 2);
  u16* f3p = (u16*)alloc((size_t)ROWS4 * 1200 * 2);
  u16* h1 = (u16*)alloc((size_t)MBP * KP * 2);
  u16* h2 = (u16*)alloc((size_t)MBP * KP * 2);
  u16* h3 = (u16*)alloc((size_t)MBP * KP * 2);
  u16* A1 = (u16*)alloc((size_t)MBP * KP * 2);
  u16* A2 = (u16*)alloc((size_t)MBP * KP * 2);
  u16* A3 = (u16*)alloc((size_t)MBP * KP * 2);
  float* gacc1 = (float*)alloc((size_t)MBP * 800 * 4);
  float* gacc2 = (float*)alloc((size_t)MBP * 800 * 4);
  float* gacc3 = (float*)alloc((size_t)MBP * 800 * 4);
  float* upd1 = (float*)alloc((size_t)MBP * NHID * 4);
  float* upd2 = (float*)alloc((size_t)MBP * NHID * 4);
  float* upd3 = (float*)alloc((size_t)MBP * NHID * 4);
  float* f2buf = (float*)alloc((size_t)MBP * 1200 * 4);
  float* f3buf = (float*)alloc((size_t)MBP * 1200 * 4);
  float* outacc = (float*)alloc((size_t)MBP * 128 * 4);
  unsigned* bar = (unsigned*)alloc(256);

  hipMemsetAsync(bar, 0, 256, stream);
  // zero the 16 pad rows of the bf16 A-operand inputs so padded M-tiles stay finite
  hipMemsetAsync(xbf  + (size_t)ROWS * KP, 0, (size_t)(ROWS4 - ROWS) * KP * 2, stream);
  hipMemsetAsync(wbf  + (size_t)ROWS * KP, 0, (size_t)(ROWS4 - ROWS) * KP * 2, stream);
  hipMemsetAsync(wfbf + (size_t)ROWS * KP, 0, (size_t)(ROWS4 - ROWS) * KP * 2, stream);

  auto packL = [&](const float* W, u16* dst, int K, int N){
    int NT = (N + 15) / 16;
    int total = NT * KT13 * 512;
    pack_kernel<<<(total + 255) / 256, 256, 0, stream>>>(W, dst, K, N, total);
  };
  packL(ifW, pk_if, 400, 400);
  packL(afW, pk_af, 400, 1200);
  packL(p2W, pk_p2, 400, 1200);  packL(a2W, pk_a2, 400, 1200);
  packL(p3W, pk_p3, 400, 1200);  packL(a3W, pk_a3, 400, 1200);
  packL(l12W, pk_l12, 400, 1200); packL(l13W, pk_l13, 400, 1200); packL(l23W, pk_l23, 400, 1200);
  packL(wur1W, pk_wur1, 400, 800); packL(wur2W, pk_wur2, 400, 800); packL(wur3W, pk_wur3, 400, 800);
  packL(u1W, pk_u1, 400, 400); packL(u2W, pk_u2, 400, 400); packL(u3W, pk_u3, 400, 400);
  packL(lo1W, pk_lo1, 400, 121); packL(lo2W, pk_lo2, 400, 121); packL(lo3W, pk_lo3, 400, 121);
  obias_kernel<<<1, 128, 0, stream>>>(lo1b, lo2b, lo3b, obias);

  {
    size_t total = (size_t)ROWS * KP;
    xproj_kernel<<<(unsigned)((total + 255) / 256), 256, 0, stream>>>(inp, lW, lb, xbf);
  }
  heads_kernel<<<(ROWS * NATT + 255) / 256, 256, 0, stream>>>(xbf, aW, ab_, bW, bb_, kW, kb_,
                                                              aexp, bexp, kex);
  kcum_kernel<<<1, 512, 0, stream>>>(kinit, kex);
  // finp = x @ inp_fork_W + b (bf16)
  gemm_pre<<<512, 256, 0, stream>>>(xbf, pk_if, nullptr, nullptr, ifb, nullptr,
                                    finp, nullptr, ROWS4, 25, 400);
  attn_kernel<<<ROWS, 128, 0, stream>>>(c_inp, emb, aexp, bexp, kex, finp, wbf, wfbf);
  // F1 = (w+finp) @ att_fork_W + b
  gemm_pre<<<512, 256, 0, stream>>>(wfbf, pk_af, nullptr, nullptr, afb, nullptr,
                                    F1, nullptr, ROWS4, 75, 1200);
  // f2p = x@proj_to_l2 + w@att_to_l2 + both biases ; f3p likewise
  gemm_pre<<<512, 256, 0, stream>>>(xbf, pk_p2, wbf, pk_a2, p2b, a2b,
                                    f2p, nullptr, ROWS4, 75, 1200);
  gemm_pre<<<512, 256, 0, stream>>>(xbf, pk_p3, wbf, pk_a3, p3b, a3b,
                                    f3p, nullptr, ROWS4, 75, 1200);

  SeqArgs S;
  S.pk_u1 = pk_u1; S.pk_u2 = pk_u2; S.pk_u3 = pk_u3;
  S.pk_wur1 = pk_wur1; S.pk_wur2 = pk_wur2; S.pk_wur3 = pk_wur3;
  S.pk_l12 = pk_l12; S.pk_l13 = pk_l13; S.pk_l23 = pk_l23;
  S.pk_lo1 = pk_lo1; S.pk_lo2 = pk_lo2; S.pk_lo3 = pk_lo3;
  S.F1 = F1; S.f2p = f2p; S.f3p = f3p;
  S.l12b = l12b; S.l13b = l13b; S.l23b = l23b;
  S.poutW = poW; S.poutB = pob; S.obias = obias; S.mask = mask;
  S.h1i = h1i; S.h2i = h2i; S.h3i = h3i;
  S.h1 = h1; S.h2 = h2; S.h3 = h3; S.A1 = A1; S.A2 = A2; S.A3 = A3;
  S.gacc1 = gacc1; S.gacc2 = gacc2; S.gacc3 = gacc3;
  S.upd1 = upd1; S.upd2 = upd2; S.upd3 = upd3;
  S.f2buf = f2buf; S.f3buf = f3buf; S.outacc = outacc;
  S.dout = (float*)d_out; S.bar = bar;
  seq_kernel<<<NWG, TPB, 0, stream>>>(S);
}
